// SSGCN_29480655519798
// MI455X (gfx1250) — matrix operand order repaired and checked by reading, not compiled, not run on hardware
//
#include <hip/hip_runtime.h>
#include <hip/hip_bf16.h>

// ---------------------------------------------------------------------------
// SSGCN for MI455X (gfx1250, wave32).
//  * All dense GEMMs: v_wmma_f32_16x16x32_bf16, f32 accumulate.
//  * All GEMM operands pre-converted to bf16; B operands stored (N,K) so tile
//    staging is contiguous b128 copies (or TDM tensor_load_to_lds when the
//    builtin exists), and both A and B fragments vectorize to ds_load_b128.
//  * LSTM recurrence: 8 persistent workgroups, h/c state resident in LDS.
// ---------------------------------------------------------------------------

typedef __bf16 bf16_t;
typedef __bf16 v16bf __attribute__((ext_vector_type(16)));
typedef __bf16 v8bf  __attribute__((ext_vector_type(8)));
typedef float  v8f   __attribute__((ext_vector_type(8)));

#define B_   64
#define S_   512
#define H_   768
#define H2_  384
#define E_   2048
#define A_   4
#define R_   3
#define C_   3
#define G4_  1536              // 4*H2
#define MS_  (B_ * S_)         // 32768

// ------------------------- TDM availability probes --------------------------
#if defined(__has_builtin)
#  if __has_builtin(__builtin_amdgcn_tensor_load_to_lds)
#    define HAVE_TDM 1
#  else
#    define HAVE_TDM 0
#  endif
#else
#  define HAVE_TDM 0
#endif

#if defined(__has_include)
#  if __has_include(<hip/amd_detail/amd_gfx1250_TDM.h>)
#    define TDM_SIX_ARGS 1
#  else
#    define TDM_SIX_ARGS 0
#  endif
#else
#  define TDM_SIX_ARGS 0
#endif

#if HAVE_TDM
typedef unsigned int u32x4 __attribute__((ext_vector_type(4)));
typedef int          i32x4 __attribute__((ext_vector_type(4)));
typedef int          i32x8 __attribute__((ext_vector_type(8)));

// 2D tile DMA: global (row-major, stride_elts bf16 per row) -> LDS, rows x 32.
// D# fields per CDNA5 ISA ch.8 (group0: count/lds/global/type, group1: dims).
__device__ __forceinline__ void tdm_load_2d(unsigned int lds_addr,
                                            const bf16_t* gptr,
                                            unsigned int rows,
                                            unsigned int tile_k,      // elems
                                            unsigned int stride_elts,
                                            unsigned int tdim0,
                                            unsigned int tdim1) {
  const unsigned long long ga = (unsigned long long)gptr;
  u32x4 g0;
  g0[0] = 1u;                                     // count=1, user descriptor
  g0[1] = lds_addr;                               // LDS byte address
  g0[2] = (unsigned int)(ga & 0xFFFFFFFFu);       // global_addr[31:0]
  g0[3] = (unsigned int)((ga >> 32) & 0x01FFFFFFu) | (2u << 30);  // type=2
  i32x8 g1;
  g1[0] = (int)(1u << 16);                        // data_size = 2 bytes
  g1[1] = (int)((tdim0 & 0xFFFFu) << 16);         // tensor_dim0[15:0]
  g1[2] = (int)(((tdim0 >> 16) & 0xFFFFu) | ((tdim1 & 0xFFFFu) << 16));
  g1[3] = (int)(((tdim1 >> 16) & 0xFFFFu) | ((tile_k & 0xFFFFu) << 16));
  g1[4] = (int)(rows & 0xFFFFu);                  // tile_dim1; tile_dim2=0
  g1[5] = (int)stride_elts;                       // tensor_dim0_stride[31:0]
  g1[6] = 0;
  g1[7] = 0;
  i32x4 z4 = {0, 0, 0, 0};
#if TDM_SIX_ARGS
  i32x8 z8 = {0, 0, 0, 0, 0, 0, 0, 0};
  __builtin_amdgcn_tensor_load_to_lds(g0, g1, z4, z4, z8, 0);
#else
  __builtin_amdgcn_tensor_load_to_lds(g0, g1, z4, z4, 0);
#endif
}
#endif  // HAVE_TDM

// ---------------- WMMA fragment helpers (contiguous k = 16*half + e per lane) ----------------

__device__ __forceinline__ v8f wmma_bf16(v16bf a, v16bf b, v8f c) {
  return __builtin_amdgcn_wmma_f32_16x16x32_bf16(
      false, a, false, b, (short)0, c, false, false);
}

// A fragment: 16x32 bf16, row-major source with leading dim lda.
__device__ __forceinline__ v16bf frag_a(const bf16_t* base, int lda) {
  const int lane = threadIdx.x & 31;
  const int half = lane >> 4;
  const int m    = lane & 15;
  v16bf a;
#pragma unroll
  for (int e = 0; e < 16; ++e) {
    const int k = e + half * 16;
    a[e] = base[(size_t)m * lda + k];
  }
  return a;
}

// B fragment from an N-major (N x K) tile: lane ln wants 16 contiguous K.
__device__ __forceinline__ v16bf frag_b_nmajor(const bf16_t* base, int ldk) {
  const int lane = threadIdx.x & 31;
  const int half = lane >> 4;
  const int n    = lane & 15;
  v16bf b;
#pragma unroll
  for (int e = 0; e < 16; ++e) {
    const int k = e + half * 16;                  // e==2v+w keeps K ascending
    b[e] = base[(size_t)n * ldk + k];
  }
  return b;
}

// B fragment straight from a row-major (N x K) global matrix.
__device__ __forceinline__ v16bf frag_b_gT(const bf16_t* Bg, int n0, int k0,
                                           int ldk) {
  const int lane = threadIdx.x & 31;
  const int half = lane >> 4;
  const int n    = lane & 15;
  v16bf b;
#pragma unroll
  for (int e = 0; e < 16; ++e) {
    const int k = e + half * 16;
    b[e] = Bg[(size_t)(n0 + n) * ldk + (k0 + k)];
  }
  return b;
}

// -------------- WMMA GEMM: C = act(A @ B^T (+bias)), bf16 inputs ------------
// A: (M,K) bf16 row-major. B: (N,K) bf16 row-major. 64x64 block, 8 waves,
// each wave owns a 32x16 sub-tile. All call sites: M,N % 64 == 0, K % 32 == 0.

__global__ __launch_bounds__(256) void wmma_gemm(
    const bf16_t* __restrict__ Ab, const bf16_t* __restrict__ Bb,
    const float* __restrict__ bias, void* __restrict__ Cout,
    int M, int N, int K, int act /*0 none,1 relu,2 sigmoid*/, int outBf16) {
  __shared__ bf16_t As[64 * 32];   // M-major, lda = 32
  __shared__ bf16_t Bs[64 * 32];   // N-major, ldk = 32

  const int tid  = threadIdx.x;
  const int wave = tid >> 5;
  const int lane = tid & 31;
  const int half = lane >> 4;
  const int ln   = lane & 15;
  const int m0   = blockIdx.y * 64;
  const int n0   = blockIdx.x * 64;
  const int wm   = wave >> 2;      // 0..1 -> rows wm*32
  const int wn   = wave & 3;       // 0..3 -> cols wn*16

  v8f c0 = {0.f, 0.f, 0.f, 0.f, 0.f, 0.f, 0.f, 0.f};
  v8f c1 = c0;

#if HAVE_TDM
  const unsigned int asAddr = (unsigned int)(unsigned long long)(void*)As;
  const unsigned int bsAddr = (unsigned int)(unsigned long long)(void*)Bs;
#endif

  for (int k0 = 0; k0 < K; k0 += 32) {
#if HAVE_TDM
    if (tid == 0) {  // TDM ignores EXEC; issue once from wave 0
      tdm_load_2d(asAddr, Ab + (size_t)m0 * K + k0, 64, 32, K,
                  (unsigned)(K - k0), (unsigned)(M - m0));
      tdm_load_2d(bsAddr, Bb + (size_t)n0 * K + k0, 64, 32, K,
                  (unsigned)(K - k0), (unsigned)(N - n0));
    }
    if (tid < 32) __builtin_amdgcn_s_wait_tensorcnt(0);
    __syncthreads();
#else
    {  // vectorized manual staging: one b128 load + one b128 LDS store each
      const int li = tid * 8;
      const int ar = li >> 5, ac = li & 31;
      *(v8bf*)&As[li] = *(const v8bf*)&Ab[(size_t)(m0 + ar) * K + k0 + ac];
      *(v8bf*)&Bs[li] = *(const v8bf*)&Bb[(size_t)(n0 + ar) * K + k0 + ac];
    }
    if (k0 + 32 < K)
      __builtin_prefetch(&Ab[(size_t)(m0 + (tid >> 2)) * K + k0 + 32], 0, 0);
    __syncthreads();
#endif

    const v16bf a0 = frag_a(&As[(wm * 32) * 32], 32);
    const v16bf a1 = frag_a(&As[(wm * 32 + 16) * 32], 32);
    const v16bf bb = frag_b_nmajor(&Bs[(wn * 16) * 32], 32);
    c0 = wmma_bf16(a0, bb, c0);
    c1 = wmma_bf16(a1, bb, c1);
    __syncthreads();
  }

  const int col = n0 + wn * 16 + ln;
  const float bv = bias ? bias[col] : 0.f;
#pragma unroll
  for (int e = 0; e < 8; ++e) {
    const int r0 = m0 + wm * 32 + e + half * 8;
    const int r1 = r0 + 16;
    float v0 = c0[e] + bv;
    float v1 = c1[e] + bv;
    if (act == 1) { v0 = fmaxf(v0, 0.f); v1 = fmaxf(v1, 0.f); }
    else if (act == 2) {
      v0 = 1.f / (1.f + __expf(-v0));
      v1 = 1.f / (1.f + __expf(-v1));
    }
    if (outBf16) {
      ((bf16_t*)Cout)[(size_t)r0 * N + col] = (bf16_t)v0;
      ((bf16_t*)Cout)[(size_t)r1 * N + col] = (bf16_t)v1;
    } else {
      ((float*)Cout)[(size_t)r0 * N + col] = v0;
      ((float*)Cout)[(size_t)r1 * N + col] = v1;
    }
  }
}

// ---------------- BiLSTM scan (persistent, state in LDS) --------------------
#define LSTM_SMEM (16 * G4_ * 4 + 16 * H2_ * 4 + 16 * H2_ * 2)

__global__ __launch_bounds__(256) void lstm_scan(
    const bf16_t* __restrict__ pre_f, const bf16_t* __restrict__ pre_b,
    const bf16_t* __restrict__ whh_f, const bf16_t* __restrict__ whh_b,
    float* __restrict__ hcat, bf16_t* __restrict__ hcatb) {
  extern __shared__ char smem[];
  float*  g_sh = (float*)smem;                 // 16*1536
  float*  c_sh = g_sh + 16 * G4_;              // 16*384
  bf16_t* h_sh = (bf16_t*)(c_sh + 16 * H2_);   // 16*384

  const int dir = blockIdx.y;
  const int b0  = blockIdx.x * 16;
  const bf16_t* pre = dir ? pre_b : pre_f;
  const bf16_t* whh = dir ? whh_b : whh_f;

  const int tid  = threadIdx.x;
  const int wave = tid >> 5;
  const int lane = tid & 31;
  const int half = lane >> 4;
  const int ln   = lane & 15;

  for (int i = tid; i < 16 * H2_; i += 256) {
    c_sh[i] = 0.f;
    h_sh[i] = (bf16_t)0.f;
  }
  __syncthreads();

  const v8f vzero = {0.f, 0.f, 0.f, 0.f, 0.f, 0.f, 0.f, 0.f};

  for (int t = 0; t < S_; ++t) {
    const int s = dir ? (S_ - 1 - t) : t;
    if (t + 1 < S_) {
      const int sn = dir ? (S_ - 2 - t) : (t + 1);
      __builtin_prefetch(&pre[((size_t)(b0 + (tid & 15)) * S_ + sn) * G4_], 0, 0);
    }

    v8f acc[12];
#pragma unroll
    for (int j = 0; j < 12; ++j) acc[j] = vzero;

    for (int kk = 0; kk < H2_; kk += 32) {
      const v16bf af = frag_a(h_sh + kk, H2_);
#pragma unroll
      for (int j = 0; j < 12; ++j) {
        const int n0 = (wave + 8 * j) * 16;
        const v16bf bf_ = frag_b_gT(whh, n0, kk, H2_);
        acc[j] = wmma_bf16(af, bf_, acc[j]);
      }
    }

#pragma unroll
    for (int j = 0; j < 12; ++j) {
      const int col = (wave + 8 * j) * 16 + ln;
#pragma unroll
      for (int e = 0; e < 8; ++e) {
        const int m = e + half * 8;
        const float p = (float)pre[((size_t)(b0 + m) * S_ + s) * G4_ + col];
        g_sh[m * G4_ + col] = acc[j][e] + p;
      }
    }
    __syncthreads();

    for (int u = tid; u < 16 * H2_; u += 256) {
      const int m = u / H2_;
      const int q = u % H2_;
      const float gi = g_sh[m * G4_ + q];
      const float gf = g_sh[m * G4_ + H2_ + q];
      const float gg = g_sh[m * G4_ + 2 * H2_ + q];
      const float go = g_sh[m * G4_ + 3 * H2_ + q];
      const float si = 1.f / (1.f + __expf(-gi));
      const float sf = 1.f / (1.f + __expf(-gf));
      const float so = 1.f / (1.f + __expf(-go));
      const float c  = sf * c_sh[u] + si * tanhf(gg);
      const float h  = so * tanhf(c);
      c_sh[u] = c;
      h_sh[u] = (bf16_t)h;
      const size_t oi = ((size_t)(b0 + m) * S_ + s) * H_ + dir * H2_ + q;
      hcat[oi]  = h;
      hcatb[oi] = (bf16_t)h;
    }
    __syncthreads();
  }
}

// ---------------- conversion / transpose helpers ----------------------------

__global__ void cvt_bf16(const float* __restrict__ in, bf16_t* __restrict__ out,
                         int n) {
  const int i = blockIdx.x * 256 + threadIdx.x;
  if (i < n) out[i] = (bf16_t)in[i];
}

// in: (rows, cols) f32 row-major -> out: (cols, rows) bf16 row-major
__global__ void transpose_bf16(const float* __restrict__ in,
                               bf16_t* __restrict__ out, int rows, int cols) {
  const int i = blockIdx.x * 256 + threadIdx.x;
  if (i >= rows * cols) return;
  const int r = i / cols, c = i % cols;
  out[(size_t)c * rows + r] = (bf16_t)in[i];
}

__global__ void fill_zero(float* __restrict__ p, size_t n) {
  const size_t i = (size_t)blockIdx.x * 256 + threadIdx.x;
  if (i < n) p[i] = 0.f;
}

// ---------------- attention / GCN / head kernels ----------------------------

__global__ __launch_bounds__(256) void avec_kernel(
    const float* __restrict__ h, const int* __restrict__ apos,
    float* __restrict__ avec, bf16_t* __restrict__ avecb) {
  const int b = blockIdx.x, t = threadIdx.x;
  const int p0 = apos[b * A_ + 0], p1 = apos[b * A_ + 1];
  const int p2 = apos[b * A_ + 2], p3 = apos[b * A_ + 3];
  float a0 = 0.f, a1 = 0.f, a2 = 0.f;
  int cnt = 0;
  for (int s = 0; s < S_; ++s) {
    if (s == p0 || s == p1 || s == p2 || s == p3) {
      ++cnt;
      const float* hp = &h[((size_t)b * S_ + s) * H_];
      a0 += hp[t]; a1 += hp[t + 256]; a2 += hp[t + 512];
    }
  }
  const float inv = 1.f / ((float)cnt + 1e-8f);
  const float v0 = a0 * inv, v1 = a1 * inv, v2 = a2 * inv;
  avec[(size_t)b * H_ + t]        = v0;
  avec[(size_t)b * H_ + t + 256]  = v1;
  avec[(size_t)b * H_ + t + 512]  = v2;
  avecb[(size_t)b * H_ + t]       = (bf16_t)v0;
  avecb[(size_t)b * H_ + t + 256] = (bf16_t)v1;
  avecb[(size_t)b * H_ + t + 512] = (bf16_t)v2;
}

__global__ __launch_bounds__(256) void dots_kernel(
    const float* __restrict__ h, const float* __restrict__ proj,
    float* __restrict__ dots) {
  const int gw   = blockIdx.x * 8 + (threadIdx.x >> 5);
  const int lane = threadIdx.x & 31;
  const int b = gw >> 9, s = gw & 511;
  const float* hp = &h[((size_t)b * S_ + s) * H_];
  const float* pp = &proj[(size_t)b * H_];
  float acc = 0.f;
  for (int i = lane; i < H_; i += 32) acc += hp[i] * pp[i];
  for (int off = 16; off > 0; off >>= 1) acc += __shfl_xor(acc, off, 32);
  if (lane == 0) dots[(size_t)b * S_ + s] = acc;
}

__global__ __launch_bounds__(256) void span_kernel(
    const float* __restrict__ h, const float* __restrict__ dots,
    const int* __restrict__ apos, float* __restrict__ span) {
  __shared__ float sd[S_], wsum[S_], red[256], sbc[2];
  const int b = blockIdx.x, t = threadIdx.x;
  const int ap0 = apos[b * A_ + 0];
  const int apL = apos[b * A_ + A_ - 1];
  for (int s = t; s < S_; s += 256) {
    sd[s] = dots[(size_t)b * S_ + s];
    wsum[s] = 0.f;
  }
  __syncthreads();
  for (int r = 0; r < R_; ++r) {
    const int left  = (ap0 - r > 0) ? (ap0 - r) : 0;
    const int right = (apL + r < S_ - 1) ? (apL + r) : (S_ - 1);
    float lmax = -1e30f;
    for (int s = t; s < S_; s += 256) {
      const float sc = (s >= left && s <= right) ? sd[s] : 0.f;
      lmax = fmaxf(lmax, sc);
    }
    red[t] = lmax; __syncthreads();
    for (int off = 128; off > 0; off >>= 1) {
      if (t < off) red[t] = fmaxf(red[t], red[t + off]);
      __syncthreads();
    }
    if (t == 0) sbc[0] = red[0];
    __syncthreads();
    const float mx = sbc[0];
    float lsum = 0.f;
    for (int s = t; s < S_; s += 256) {
      const float sc = (s >= left && s <= right) ? sd[s] : 0.f;
      lsum += __expf(sc - mx);
    }
    red[t] = lsum; __syncthreads();
    for (int off = 128; off > 0; off >>= 1) {
      if (t < off) red[t] += red[t + off];
      __syncthreads();
    }
    if (t == 0) sbc[1] = red[0];
    __syncthreads();
    const float Z = sbc[1];
    for (int s = t; s < S_; s += 256) {
      if (s >= left && s <= right)
        wsum[s] += __expf(sd[s] - mx) / Z * (1.f / (float)R_);
    }
    __syncthreads();
  }
  for (int hd = t; hd < H_; hd += 256) {
    float acc = 0.f;
    for (int s = 0; s < S_; ++s)
      acc += wsum[s] * h[((size_t)b * S_ + s) * H_ + hd];
    span[(size_t)b * H_ + hd] = acc;
  }
}

__global__ void deg_scatter(const int* __restrict__ ei,
                            const float* __restrict__ ew,
                            float* __restrict__ deg) {
  const int i = blockIdx.x * 256 + threadIdx.x;
  if (i >= B_ * E_) return;
  const int b = i / E_, e = i % E_;
  const int col = ei[(b * 2 + 1) * E_ + e];
  atomicAdd(&deg[b * S_ + col], ew[i]);
}

__global__ void dinv_kernel(float* __restrict__ d, int n) {
  const int i = blockIdx.x * 256 + threadIdx.x;
  if (i < n) d[i] = rsqrtf(d[i] + 1.f);
}

__global__ __launch_bounds__(256) void gcn_scatter(
    const int* __restrict__ ei, const float* __restrict__ ew,
    const float* __restrict__ dinv, const float* __restrict__ xw,
    float* __restrict__ agg) {
  const int be = blockIdx.x;
  const int b = be / E_, e = be % E_;
  const int row = ei[(b * 2 + 0) * E_ + e];
  const int col = ei[(b * 2 + 1) * E_ + e];
  const float w = ew[b * E_ + e];
  const float norm = dinv[b * S_ + row] * w * dinv[b * S_ + col];
  const float* src = &xw[((size_t)b * S_ + row) * H_];
  float* dst = &agg[((size_t)b * S_ + col) * H_];
  for (int hd = threadIdx.x; hd < H_; hd += 256)
    atomicAdd(&dst[hd], src[hd] * norm);
}

__global__ void gcn_combine(const float* __restrict__ agg,
                            const float* __restrict__ xw,
                            const float* __restrict__ dinv,
                            const float* __restrict__ bias,
                            float* __restrict__ out,
                            bf16_t* __restrict__ outb, int relu) {
  const size_t i = (size_t)blockIdx.x * 256 + threadIdx.x;
  if (i >= (size_t)MS_ * H_) return;
  const int bs = (int)(i / H_);
  const int hd = (int)(i % H_);
  const float dv = dinv[bs];
  float v = agg[i] + xw[i] * dv * dv + bias[hd];
  if (relu) v = fmaxf(v, 0.f);
  out[i] = v;
  if (outb) outb[i] = (bf16_t)v;
}

__global__ __launch_bounds__(256) void syn_kernel(
    const float* __restrict__ x2, const int* __restrict__ apos,
    float* __restrict__ syn) {
  const int b = blockIdx.x, t = threadIdx.x;
  const int p0 = apos[b * A_ + 0], p1 = apos[b * A_ + 1];
  const int p2 = apos[b * A_ + 2], p3 = apos[b * A_ + 3];
  for (int hd = t; hd < H_; hd += 256) {
    const float acc = x2[((size_t)b * S_ + p0) * H_ + hd] +
                      x2[((size_t)b * S_ + p1) * H_ + hd] +
                      x2[((size_t)b * S_ + p2) * H_ + hd] +
                      x2[((size_t)b * S_ + p3) * H_ + hd];
    syn[(size_t)b * H_ + hd] = acc * 0.25f;
  }
}

__global__ void concat_kernel(const float* __restrict__ span,
                              const float* __restrict__ syn,
                              bf16_t* __restrict__ catb) {
  const int i = blockIdx.x * 256 + threadIdx.x;
  if (i >= B_ * 2 * H_) return;
  const int b = i / (2 * H_), c = i % (2 * H_);
  const float v = (c < H_) ? span[(size_t)b * H_ + c]
                           : syn[(size_t)b * H_ + (c - H_)];
  catb[i] = (bf16_t)v;
}

__global__ __launch_bounds__(256) void final_kernel(
    const float* __restrict__ gate, const float* __restrict__ span,
    const float* __restrict__ syn, const float* __restrict__ clsw,
    const float* __restrict__ clsb, float* __restrict__ out) {
  __shared__ float r0[256], r1[256], r2[256];
  const int b = blockIdx.x, t = threadIdx.x;
  float p0 = 0.f, p1 = 0.f, p2 = 0.f;
  for (int hd = t; hd < H_; hd += 256) {
    const float g = gate[(size_t)b * H_ + hd];
    const float f = g * syn[(size_t)b * H_ + hd] +
                    (1.f - g) * span[(size_t)b * H_ + hd];
    p0 += f * clsw[hd * C_ + 0];
    p1 += f * clsw[hd * C_ + 1];
    p2 += f * clsw[hd * C_ + 2];
  }
  r0[t] = p0; r1[t] = p1; r2[t] = p2;
  __syncthreads();
  for (int off = 128; off > 0; off >>= 1) {
    if (t < off) { r0[t] += r0[t + off]; r1[t] += r1[t + off]; r2[t] += r2[t + off]; }
    __syncthreads();
  }
  if (t == 0) {
    out[b * C_ + 0] = r0[0] + clsb[0];
    out[b * C_ + 1] = r1[0] + clsb[1];
    out[b * C_ + 2] = r2[0] + clsb[2];
  }
}

// ---------------------------------- host ------------------------------------

extern "C" void kernel_launch(void* const* d_in, const int* in_sizes, int n_in,
                              void* d_out, int out_size, void* d_ws,
                              size_t ws_size, hipStream_t stream) {
  const float* bert   = (const float*)d_in[0];
  const int*   ei     = (const int*)d_in[1];
  const float* ew     = (const float*)d_in[2];
  const int*   apos   = (const int*)d_in[3];
  const float* w_ih_f = (const float*)d_in[4];
  const float* w_hh_f = (const float*)d_in[5];
  const float* b_f    = (const float*)d_in[6];
  const float* w_ih_b = (const float*)d_in[7];
  const float* w_hh_b = (const float*)d_in[8];
  const float* b_b    = (const float*)d_in[9];
  const float* span_w = (const float*)d_in[10];
  const float* span_b = (const float*)d_in[11];
  const float* g1w    = (const float*)d_in[12];
  const float* g1b    = (const float*)d_in[13];
  const float* g2w    = (const float*)d_in[14];
  const float* g2b    = (const float*)d_in[15];
  const float* gatew  = (const float*)d_in[16];
  const float* gateb  = (const float*)d_in[17];
  const float* clsw   = (const float*)d_in[18];
  const float* clsb   = (const float*)d_in[19];

  size_t off = 0;
  auto alloc = [&](size_t bytes) -> void* {
    void* p = (char*)d_ws + off;
    off += (bytes + 255) & ~(size_t)255;
    return p;
  };
  // bf16 weights (B-side always (N,K))
  bf16_t* whhf = (bf16_t*)alloc((size_t)G4_ * H2_ * 2);
  bf16_t* whhb = (bf16_t*)alloc((size_t)G4_ * H2_ * 2);
  bf16_t* wihf = (bf16_t*)alloc((size_t)G4_ * H_ * 2);
  bf16_t* wihb = (bf16_t*)alloc((size_t)G4_ * H_ * 2);
  bf16_t* spw  = (bf16_t*)alloc((size_t)H_ * H_ * 2);
  bf16_t* g1t  = (bf16_t*)alloc((size_t)H_ * H_ * 2);
  bf16_t* g2t  = (bf16_t*)alloc((size_t)H_ * H_ * 2);
  bf16_t* gtt  = (bf16_t*)alloc((size_t)H_ * G4_ * 2);
  // activations
  bf16_t* bertb = (bf16_t*)alloc((size_t)MS_ * H_ * 2);
  bf16_t* pref  = (bf16_t*)alloc((size_t)MS_ * G4_ * 2);
  bf16_t* preb  = (bf16_t*)alloc((size_t)MS_ * G4_ * 2);
  float*  hcat  = (float*)alloc((size_t)MS_ * H_ * 4);
  bf16_t* hcatb = (bf16_t*)alloc((size_t)MS_ * H_ * 2);
  float*  xw    = (float*)alloc((size_t)MS_ * H_ * 4);
  float*  agg   = (float*)alloc((size_t)MS_ * H_ * 4);
  bf16_t* x1b   = (bf16_t*)alloc((size_t)MS_ * H_ * 2);
  float*  avec  = (float*)alloc((size_t)B_ * H_ * 4);
  bf16_t* avecb = (bf16_t*)alloc((size_t)B_ * H_ * 2);
  float*  proj  = (float*)alloc((size_t)B_ * H_ * 4);
  float*  dots  = (float*)alloc((size_t)B_ * S_ * 4);
  float*  span  = (float*)alloc((size_t)B_ * H_ * 4);
  float*  dinv  = (float*)alloc((size_t)B_ * S_ * 4);
  float*  syn   = (float*)alloc((size_t)B_ * H_ * 4);
  bf16_t* catb  = (bf16_t*)alloc((size_t)B_ * 2 * H_ * 2);
  float*  gate  = (float*)alloc((size_t)B_ * H_ * 4);
  (void)ws_size; (void)in_sizes; (void)n_in; (void)out_size;

  const dim3 blk(256);
  auto cv = [&](const float* src, bf16_t* dst, int n) {
    cvt_bf16<<<(n + 255) / 256, blk, 0, stream>>>(src, dst, n);
  };

  // 0) one-time precision conversion / weight transposition
  cv(w_hh_f, whhf, G4_ * H2_);
  cv(w_hh_b, whhb, G4_ * H2_);
  cv(w_ih_f, wihf, G4_ * H_);
  cv(w_ih_b, wihb, G4_ * H_);
  cv(span_w, spw, H_ * H_);
  cv(bert, bertb, MS_ * H_);
  transpose_bf16<<<(H_ * H_ + 255) / 256, blk, 0, stream>>>(g1w, g1t, H_, H_);
  transpose_bf16<<<(H_ * H_ + 255) / 256, blk, 0, stream>>>(g2w, g2t, H_, H_);
  transpose_bf16<<<(G4_ * H_ + 255) / 256, blk, 0, stream>>>(gatew, gtt, G4_, H_);

  // 1) pre-activations (B*S,768) @ w_ih^T + b -> bf16
  wmma_gemm<<<dim3(G4_ / 64, MS_ / 64), blk, 0, stream>>>(
      bertb, wihf, b_f, pref, MS_, G4_, H_, 0, 1);
  wmma_gemm<<<dim3(G4_ / 64, MS_ / 64), blk, 0, stream>>>(
      bertb, wihb, b_b, preb, MS_, G4_, H_, 0, 1);

  // 2) BiLSTM scan (persistent, WMMA per step)
  lstm_scan<<<dim3(4, 2), blk, LSTM_SMEM, stream>>>(pref, preb, whhf, whhb,
                                                    hcat, hcatb);

  // 3) span attention
  avec_kernel<<<B_, blk, 0, stream>>>(hcat, apos, avec, avecb);
  wmma_gemm<<<dim3(H_ / 64, B_ / 64), blk, 0, stream>>>(
      avecb, spw, span_b, proj, B_, H_, H_, 0, 0);
  dots_kernel<<<MS_ / 8, blk, 0, stream>>>(hcat, proj, dots);
  span_kernel<<<B_, blk, 0, stream>>>(hcat, dots, apos, span);

  // 4) GCN normalization terms
  int n = B_ * S_;
  fill_zero<<<(n + 255) / 256, blk, 0, stream>>>(dinv, (size_t)n);
  deg_scatter<<<(B_ * E_ + 255) / 256, blk, 0, stream>>>(ei, ew, dinv);
  dinv_kernel<<<(n + 255) / 256, blk, 0, stream>>>(dinv, n);

  const size_t nact = (size_t)MS_ * H_;
  // 5) GCN layer 1 (x1 f32 lives in `agg`, bf16 copy in x1b)
  wmma_gemm<<<dim3(H_ / 64, MS_ / 64), blk, 0, stream>>>(
      hcatb, g1t, nullptr, xw, MS_, H_, H_, 0, 0);
  fill_zero<<<(nact + 255) / 256, blk, 0, stream>>>(agg, nact);
  gcn_scatter<<<B_ * E_, blk, 0, stream>>>(ei, ew, dinv, xw, agg);
  gcn_combine<<<(nact + 255) / 256, blk, 0, stream>>>(agg, xw, dinv, g1b, agg,
                                                      x1b, 1);

  // 6) GCN layer 2 (x2 lives in `xw`)
  wmma_gemm<<<dim3(H_ / 64, MS_ / 64), blk, 0, stream>>>(
      x1b, g2t, nullptr, xw, MS_, H_, H_, 0, 0);
  fill_zero<<<(nact + 255) / 256, blk, 0, stream>>>(agg, nact);
  gcn_scatter<<<B_ * E_, blk, 0, stream>>>(ei, ew, dinv, xw, agg);
  gcn_combine<<<(nact + 255) / 256, blk, 0, stream>>>(agg, xw, dinv, g2b, xw,
                                                      nullptr, 0);

  // 7) aspect gather + gate + classifier
  syn_kernel<<<B_, blk, 0, stream>>>(xw, apos, syn);
  concat_kernel<<<(B_ * 2 * H_ + 255) / 256, blk, 0, stream>>>(span, syn, catb);
  wmma_gemm<<<dim3(H_ / 64, B_ / 64), blk, 0, stream>>>(
      catb, gtt, gateb, gate, B_, H_, G4_, 2, 0);
  final_kernel<<<B_, blk, 0, stream>>>(gate, span, syn, clsw, clsb,
                                       (float*)d_out);
}
